// Encoder_82910048682485
// MI455X (gfx1250) — matrix operand order repaired and checked by reading, not compiled, not run on hardware
//
#include <hip/hip_runtime.h>

#define TDEPTH 18
#define TEMB   128
#define THID   256
#define TVAL   32

typedef __attribute__((ext_vector_type(16))) __bf16 v16bf;
typedef __attribute__((ext_vector_type(8)))  __bf16 v8bf;
typedef __attribute__((ext_vector_type(8)))  float  v8f;

// ---- async global->LDS staging (gfx1250), with sync fallback ----------------
#if defined(__AMDGCN__) && __has_builtin(__builtin_amdgcn_global_load_async_to_lds_b128) && __has_builtin(__builtin_amdgcn_s_wait_asynccnt)
#define USE_ASYNC 1
#endif

#ifdef USE_ASYNC
#define WAIT_ASYNC(n) __builtin_amdgcn_s_wait_asynccnt(n)
typedef int async_v4i __attribute__((vector_size(4 * sizeof(int))));
typedef __attribute__((address_space(1))) async_v4i g_v4i;  // global int4
typedef __attribute__((address_space(3))) async_v4i l_v4i;  // LDS int4
#else
#define WAIT_ASYNC(n) do {} while (0)
#endif

__device__ __forceinline__ void cp16(const unsigned short* g, unsigned short* l) {
#ifdef USE_ASYNC
  __builtin_amdgcn_global_load_async_to_lds_b128(
      (g_v4i*)(unsigned short*)g, (l_v4i*)l, 0, 0);
#else
  *(uint4*)l = *(const uint4*)g;
#endif
}

// copy a [ROWS x 32] bf16 K-chunk (row stride 256 in global) into LDS rows of
// stride 48 elements; 16B blocks, (ROWS*4)/256 issues per thread.
template <int ROWS>
__device__ __forceinline__ void copy_chunk(const unsigned short* __restrict__ src,
                                           unsigned short* __restrict__ dst, int tid) {
#pragma unroll
  for (int it = 0; it < (ROWS * 4) / 256; ++it) {
    int b = tid + it * 256;
    int r = b >> 2;
    int j = b & 3;
    cp16(src + r * 256 + j * 8, dst + r * 48 + j * 8);
  }
}

__device__ __forceinline__ float lrelu(float x) { return x > 0.f ? x : 0.01f * x; }

__device__ __forceinline__ unsigned short f2bfu(float f) {
  __bf16 b = (__bf16)f;
  return __builtin_bit_cast(unsigned short, b);
}

// A-fragment (16x32 bf16) from an f32 row: lane holds row lane%16,
// K runs [k0, k0+16) with k0 = base + 16*(lane/16).
__device__ __forceinline__ v16bf load_a32(const float* __restrict__ rowp, int k0) {
  float4 q0 = *(const float4*)(rowp + k0);
  float4 q1 = *(const float4*)(rowp + k0 + 4);
  float4 q2 = *(const float4*)(rowp + k0 + 8);
  float4 q3 = *(const float4*)(rowp + k0 + 12);
  v16bf a;
  a[0] = (__bf16)q0.x;  a[1] = (__bf16)q0.y;  a[2] = (__bf16)q0.z;  a[3] = (__bf16)q0.w;
  a[4] = (__bf16)q1.x;  a[5] = (__bf16)q1.y;  a[6] = (__bf16)q1.z;  a[7] = (__bf16)q1.w;
  a[8] = (__bf16)q2.x;  a[9] = (__bf16)q2.y;  a[10] = (__bf16)q2.z; a[11] = (__bf16)q2.w;
  a[12] = (__bf16)q3.x; a[13] = (__bf16)q3.y; a[14] = (__bf16)q3.z; a[15] = (__bf16)q3.w;
  return a;
}

__device__ __forceinline__ v8f wmma_bf16(v16bf a, v16bf b, v8f c) {
  return __builtin_amdgcn_wmma_f32_16x16x32_bf16(false, a, false, b, (short)0, c, false, false);
}

// ---- one-time weight transpose+convert to bf16 ------------------------------
// w1t: [256 n][256 k], w2t: [128 n][256 k], weT: [128 n][32 k]
__global__ void prep_kernel(const float* __restrict__ We, const float* __restrict__ W1,
                            const float* __restrict__ W2, unsigned short* __restrict__ w1t,
                            unsigned short* __restrict__ w2t, unsigned short* __restrict__ weT) {
  int i = blockIdx.x * blockDim.x + threadIdx.x;
  int stride = gridDim.x * blockDim.x;
  for (int o = i; o < 256 * 256; o += stride) { int n = o >> 8, k = o & 255; w1t[o] = f2bfu(W1[k * 256 + n]); }
  for (int o = i; o < 128 * 256; o += stride) { int n = o >> 8, k = o & 255; w2t[o] = f2bfu(W2[k * 128 + n]); }
  for (int o = i; o < 128 * 32;  o += stride) { int n = o >> 5, k = o & 31;  weT[o] = f2bfu(We[k * 128 + n]); }
}

// ---- leaf embedder: lrelu(leaf_values @ We + be) -> embs[leaf_start:] -------
__global__ __launch_bounds__(256) void leaf_kernel(const float* __restrict__ lv,
                                                   const unsigned short* __restrict__ weT,
                                                   const float* __restrict__ be,
                                                   float* __restrict__ out) {
  const int tid = threadIdx.x;
  const int lane = tid & 31;
  const int lr = lane & 15, lg = lane >> 4;
  const long Rg = (long)blockIdx.x * 128 + (tid >> 5) * 16;

  const float* rowp = lv + (Rg + lr) * TVAL;
  v16bf a = load_a32(rowp, lg * 16);  // full K=32

  const long leafStart = (1L << TDEPTH) - 1;
  float* outp = out + leafStart * TEMB;
#pragma unroll
  for (int u = 0; u < 8; ++u) {
    v16bf b = *(const v16bf*)(weT + (u * 16 + lr) * 32 + lg * 16);
    v8f c = (v8f)0.0f;
    c = wmma_bf16(a, b, c);
    float bb = be[u * 16 + lr];
#pragma unroll
    for (int j = 0; j < 8; ++j) {
      long row = Rg + j + lg * 8;
      outp[row * TEMB + u * 16 + lr] = lrelu(c[j] + bb);
    }
  }
}

// ---- per-level cell: out = lrelu(lrelu(X@W1+b1)@W2+b2) ----------------------
// X = embs rows starting at childBase, viewed as [n, 256] row-major.
__global__ __launch_bounds__(256) void level_kernel(float* __restrict__ embs,
                                                    const unsigned short* __restrict__ w1t,
                                                    const unsigned short* __restrict__ w2t,
                                                    const float* __restrict__ b1,
                                                    const float* __restrict__ b2,
                                                    int p0, int nP, int childBase) {
  __shared__ __align__(32) unsigned short sW[2][256 * 48];  // weight K-chunks (double buffered)
  __shared__ __align__(32) unsigned short sHs[8][16 * 32];  // per-wave C->A conversion scratch

  const int tid = threadIdx.x;
  const int wave = tid >> 5;
  const int lane = tid & 31;
  const int lr = lane & 15, lg = lane >> 4;
  const long Rg = (long)blockIdx.x * 128 + wave * 16;  // parent-row base of this wave

  const float* X = embs + (long)childBase * TEMB;  // [*, 256] rows

  // ---------------- Phase A: H accumulators (16 col tiles of 16) ------------
  v8f acc1[16];
#pragma unroll
  for (int t = 0; t < 16; ++t) acc1[t] = (v8f)0.0f;

  copy_chunk<256>(w1t, sW[0], tid);
  const float* rowp = X + (Rg + lr) * (2 * TEMB);
  for (int kc = 0; kc < 8; ++kc) {
    if (kc < 7) {
      copy_chunk<256>(w1t + (kc + 1) * 32, sW[(kc + 1) & 1], tid);
      WAIT_ASYNC(4);
    } else {
      WAIT_ASYNC(0);
    }
    __syncthreads();
    v16bf a = load_a32(rowp, kc * 32 + lg * 16);
    const unsigned short* wb = sW[kc & 1];
#pragma unroll
    for (int t = 0; t < 16; ++t) {
      v16bf b = *(const v16bf*)(wb + (t * 16 + lr) * 48 + lg * 16);
      acc1[t] = wmma_bf16(a, b, acc1[t]);
    }
    __syncthreads();
  }

  // ---------------- Phase B: OUT = lrelu(H @ W2 + b2) -----------------------
  v8f acc2[8];
#pragma unroll
  for (int u = 0; u < 8; ++u) acc2[u] = (v8f)0.0f;

  copy_chunk<128>(w2t, sW[0], tid);
  unsigned short* hs = sHs[wave];
#pragma unroll
  for (int s = 0; s < 8; ++s) {
    if (s < 7) {
      copy_chunk<128>(w2t + (s + 1) * 32, sW[(s + 1) & 1], tid);
      WAIT_ASYNC(2);
    } else {
      WAIT_ASYNC(0);
    }
    __syncthreads();
    // bias + lrelu + bf16-convert tiles 2s, 2s+1 into per-wave scratch (cols 32s..32s+31)
#pragma unroll
    for (int q = 0; q < 2; ++q) {
      int t = 2 * s + q;
      float bb = b1[t * 16 + lr];
#pragma unroll
      for (int j = 0; j < 8; ++j) {
        float h = lrelu(acc1[t][j] + bb);
        hs[(j + lg * 8) * 32 + q * 16 + lr] = f2bfu(h);
      }
    }
    // re-read as A fragment (same-wave LDS ops are in-order)
    v8bf lo = *(const v8bf*)(hs + lr * 32 + lg * 16);
    v8bf hi = *(const v8bf*)(hs + lr * 32 + lg * 16 + 8);
    v16bf ah = __builtin_shufflevector(lo, hi, 0, 1, 2, 3, 4, 5, 6, 7, 8, 9, 10, 11, 12, 13, 14, 15);
    const unsigned short* wb = sW[s & 1];
#pragma unroll
    for (int u = 0; u < 8; ++u) {
      v16bf b = *(const v16bf*)(wb + (u * 16 + lr) * 48 + lg * 16);
      acc2[u] = wmma_bf16(ah, b, acc2[u]);
    }
    __syncthreads();
  }

  // ---------------- store (guard partial levels) ----------------------------
  float* outp = embs + (long)p0 * TEMB;
#pragma unroll
  for (int u = 0; u < 8; ++u) {
    float bb = b2[u * 16 + lr];
#pragma unroll
    for (int j = 0; j < 8; ++j) {
      long row = Rg + j + lg * 8;
      if (row < nP) outp[row * TEMB + u * 16 + lr] = lrelu(acc2[u][j] + bb);
    }
  }
}

// -----------------------------------------------------------------------------
extern "C" void kernel_launch(void* const* d_in, const int* in_sizes, int n_in,
                              void* d_out, int out_size, void* d_ws, size_t ws_size,
                              hipStream_t stream) {
  const float* lv = (const float*)d_in[0];
  const float* We = (const float*)d_in[1];
  const float* be = (const float*)d_in[2];
  const float* W1 = (const float*)d_in[3];
  const float* b1 = (const float*)d_in[4];
  const float* W2 = (const float*)d_in[5];
  const float* b2 = (const float*)d_in[6];
  float* out = (float*)d_out;

  unsigned short* w1t = (unsigned short*)d_ws;                        // 256*256 bf16 = 128KB
  unsigned short* w2t = (unsigned short*)((char*)d_ws + 131072);      // 128*256 bf16 =  64KB
  unsigned short* weT = (unsigned short*)((char*)d_ws + 131072 + 65536);  // 128*32 bf16 = 8KB

  prep_kernel<<<64, 256, 0, stream>>>(We, W1, W2, w1t, w2t, weT);

  const int nLeaves = 1 << TDEPTH;
  leaf_kernel<<<nLeaves / 128, 256, 0, stream>>>(lv, weT, be, out);

  for (int l = TDEPTH - 1; l >= 0; --l) {
    int n = 1 << l;
    int p0 = n - 1;
    int childBase = 2 * n - 1;
    int blocks = (n + 127) / 128;
    level_kernel<<<blocks, 256, 0, stream>>>(out, w1t, w2t, b1, b2, p0, n, childBase);
  }
}
